// Decoder_47536698032246
// MI455X (gfx1250) — compile-verified, matrix operand order checked by reading, not run on hardware
//
#include <hip/hip_runtime.h>
#include <hip/hip_bf16.h>

// ---------------------------------------------------------------------------
// MI455X (gfx1250) attention-LSTM decoder.
// - bf16 operands + f32 WMMA accumulation (v_wmma_f32_16x16x32_bf16).
// - bf16 batch_H (79.7MB) + bf16 H_proj (67MB) = 146.7MB < 192MB L2, so the
//   121-step scan streams entirely from L2 instead of 27GB of HBM traffic.
// - Big H_proj GEMM uses the Tensor Data Mover: one tensor_load_to_lds per
//   64x32 A/B tile per k-step (TENSORcnt + s_wait_tensorcnt), fragments then
//   come from LDS (ds_load_b128), removing the 2x/4x duplicated global reads.
// ---------------------------------------------------------------------------

typedef __bf16 bf16_t;
typedef __attribute__((ext_vector_type(16))) __bf16 v16bf;
typedef __attribute__((ext_vector_type(8)))  __bf16 v8bf;
typedef __attribute__((ext_vector_type(8)))  float  v8f;
typedef __attribute__((ext_vector_type(4)))  unsigned int v4u;
typedef __attribute__((ext_vector_type(8)))  int v8i;
typedef __attribute__((ext_vector_type(4)))  int v4i;

#define Bdim 128
#define Sdim 512
#define Vdim 600
#define Hdim 512
#define Tdim 121
#define Vpad 608   // 600 -> 19*32
#define G4H  2048  // 4*H

// ---- fragment load -------------------------------------------------------
// 16-bit A/B matrix layout (ISA 7.12.2): lane L -> idx=L&15 (row for A /
// col for B), half=L>>4; element e -> K = 16*(e/8) + 8*half + (e%8).
// With p already offset by half*8, that is two contiguous 8-element runs.
__device__ __forceinline__ v16bf load_frag(const bf16_t* p) {
    union { v16bf v; v8bf h[2]; } u;
    u.h[0] = *(const v8bf*)(p);
    u.h[1] = *(const v8bf*)(p + 16);
    return u.v;
}

// ---- TDM descriptor builders (CDNA5 ISA) -------
// Group 0: count=1 | lds_addr[63:32] | global_addr[120:64] | type=2[127:126]
__device__ __forceinline__ v4u tdm_group0(unsigned lds_addr, unsigned long long gaddr) {
    v4u g;
    g[0] = 1u;                                          // count=1 (user D#)
    g[1] = lds_addr;                                    // LDS byte address
    g[2] = (unsigned)(gaddr & 0xffffffffull);           // global addr [31:0]
    g[3] = (unsigned)((gaddr >> 32) & 0x1ffffffull)     // global addr [56:32]
         | (2u << 30);                                  // type = 2 ("image")
    return g;
}
// Group 1: data_size=2B, 2-D tile: tensor_dim0/1, tile_dim0/1, dim0 stride.
__device__ __forceinline__ v8i tdm_group1(unsigned td0, unsigned td1,
                                          unsigned tile0, unsigned tile1,
                                          unsigned long long stride0) {
    v8i g = {};
    g[0] = (int)(1u << 16);                             // data_size=1 -> 2 bytes
    g[1] = (int)((td0 & 0xffffu) << 16);                // tensor_dim0 [15:0]
    g[2] = (int)(((td0 >> 16) & 0xffffu) | ((td1 & 0xffffu) << 16));
    g[3] = (int)(((td1 >> 16) & 0xffffu) | ((tile0 & 0xffffu) << 16));
    g[4] = (int)(tile1 & 0xffffu);                      // tile_dim1 (tile_dim2=0)
    g[5] = (int)(stride0 & 0xffffffffull);              // tensor_dim0_stride lo
    g[6] = (int)((stride0 >> 32) & 0xffffull);          // stride hi (dim1_stride=0)
    return g;
}

// ---- unified WMMA GEMM (direct global fragment loads) --------------------
// C[M,Nreal] = A1[M,K1] * B1[N,K1]^T (+ A2[M,K2] * B2[N,K2]^T) + bias1 + bias2
// A row-major [M,K], B row-major [N,K] (weight layout). M multiple of 64.
// Workgroup = 256 threads = 8 waves; tile 64(M) x 64(N); wave -> 16x32 of C.
__global__ void __launch_bounds__(256)
gemm_bf16_kernel(const bf16_t* __restrict__ A1, const bf16_t* __restrict__ B1, int K1,
                 const bf16_t* __restrict__ A2, const bf16_t* __restrict__ B2, int K2,
                 float* __restrict__ Cf, bf16_t* __restrict__ Cb,
                 const float* __restrict__ bias1, const float* __restrict__ bias2,
                 int Nreal)
{
    const int lane  = threadIdx.x & 31;
    const int wave  = threadIdx.x >> 5;
    const int hl    = lane >> 4;    // 0/1
    const int idx16 = lane & 15;

    const int m_base = blockIdx.y * 64 + (wave & 3) * 16;
    const int n_base = blockIdx.x * 64 + (wave >> 2) * 32;

    v8f acc0 = {};
    v8f acc1 = {};

    const bf16_t* As[2] = {A1, A2};
    const bf16_t* Bs[2] = {B1, B2};
    const int     Ks[2] = {K1, K2};

    for (int p = 0; p < 2; ++p) {
        if (!As[p]) break;
        const int K = Ks[p];
        const bf16_t* arow  = As[p] + (size_t)(m_base + idx16)      * K + hl * 8;
        const bf16_t* brow0 = Bs[p] + (size_t)(n_base + idx16)      * K + hl * 8;
        const bf16_t* brow1 = Bs[p] + (size_t)(n_base + 16 + idx16) * K + hl * 8;
        for (int k0 = 0; k0 < K; k0 += 32) {
            v16bf a  = load_frag(arow  + k0);
            v16bf b0 = load_frag(brow0 + k0);
            v16bf b1 = load_frag(brow1 + k0);
            acc0 = __builtin_amdgcn_wmma_f32_16x16x32_bf16(false, a, false, b0,
                                                           (short)0, acc0, false, false);
            acc1 = __builtin_amdgcn_wmma_f32_16x16x32_bf16(false, a, false, b1,
                                                           (short)0, acc1, false, false);
        }
    }

    // epilogue: C layout -> lane: n = n_base(+16) + idx16; VGPR r: m = 8*hl + r
    const int n0 = n_base + idx16;
    const int n1 = n_base + 16 + idx16;
    float bb0 = 0.f, bb1 = 0.f;
    if (n0 < Nreal) { if (bias1) bb0 += bias1[n0]; if (bias2) bb0 += bias2[n0]; }
    if (n1 < Nreal) { if (bias1) bb1 += bias1[n1]; if (bias2) bb1 += bias2[n1]; }

#pragma unroll
    for (int r = 0; r < 8; ++r) {
        const int m = m_base + 8 * hl + r;
        if (n0 < Nreal) {
            float v = acc0[r] + bb0;
            if (Cf) Cf[(size_t)m * Nreal + n0] = v;
            if (Cb) Cb[(size_t)m * Nreal + n0] = (bf16_t)v;
        }
        if (n1 < Nreal) {
            float v = acc1[r] + bb1;
            if (Cf) Cf[(size_t)m * Nreal + n1] = v;
            if (Cb) Cb[(size_t)m * Nreal + n1] = (bf16_t)v;
        }
    }
}

// ---- TDM-staged WMMA GEMM for the big H_proj GEMM ------------------------
// C[M,N](bf16) = A[M,K] * B[N,K]^T. 64x64 block; per k-step the Tensor Data
// Mover DMAs a 64x32 A tile and a 64x32 B tile into LDS (one issue per
// workgroup, TENSORcnt-tracked), then all 8 waves read fragments from LDS.
__global__ void __launch_bounds__(256)
gemm_bf16_tdm_kernel(const bf16_t* __restrict__ A, const bf16_t* __restrict__ B, int K,
                     bf16_t* __restrict__ Cb, int N, int Mtot, int Ntot)
{
    __shared__ alignas(16) bf16_t ldsA[64 * 32];
    __shared__ alignas(16) bf16_t ldsB[64 * 32];

    const int lane  = threadIdx.x & 31;
    const int wave  = threadIdx.x >> 5;
    const int hl    = lane >> 4;
    const int idx16 = lane & 15;

    const int m_blk = blockIdx.y * 64;
    const int n_blk = blockIdx.x * 64;

    const int a_row  = (wave & 3) * 16 + idx16;   // LDS A-tile row
    const int b_row0 = (wave >> 2) * 32 + idx16;  // LDS B-tile rows
    const int b_row1 = b_row0 + 16;

    v8f acc0 = {}, acc1 = {};

    for (int k0 = 0; k0 < K; k0 += 32) {
#if __has_builtin(__builtin_amdgcn_tensor_load_to_lds)
        if (threadIdx.x == 0) {
            unsigned long long ga = (unsigned long long)(const char*)A
                                  + ((unsigned long long)m_blk * K + k0) * 2ull;
            unsigned long long gb = (unsigned long long)(const char*)B
                                  + ((unsigned long long)n_blk * K + k0) * 2ull;
            v4u g0a = tdm_group0((unsigned)(unsigned long long)(void*)ldsA, ga);
            v4u g0b = tdm_group0((unsigned)(unsigned long long)(void*)ldsB, gb);
            v8i g1a = tdm_group1((unsigned)K, (unsigned)Mtot, 32u, 64u,
                                 (unsigned long long)K);
            v8i g1b = tdm_group1((unsigned)K, (unsigned)Ntot, 32u, 64u,
                                 (unsigned long long)K);
            v4i z4 = {};
            v8i z8 = {};
            __builtin_amdgcn_tensor_load_to_lds(g0a, g1a, z4, z4, z8, 0);
            __builtin_amdgcn_tensor_load_to_lds(g0b, g1b, z4, z4, z8, 0);
            __builtin_amdgcn_s_wait_tensorcnt((short)0);
        }
#else
        {   // cooperative fallback: 256 threads x 8 bf16 = one 64x32 tile each
            const int row = threadIdx.x >> 2;
            const int col = (threadIdx.x & 3) * 8;
            *(v8bf*)&ldsA[row * 32 + col] =
                *(const v8bf*)(A + (size_t)(m_blk + row) * K + k0 + col);
            *(v8bf*)&ldsB[row * 32 + col] =
                *(const v8bf*)(B + (size_t)(n_blk + row) * K + k0 + col);
        }
#endif
        __syncthreads();
        v16bf a  = load_frag(&ldsA[a_row  * 32 + 8 * hl]);
        v16bf b0 = load_frag(&ldsB[b_row0 * 32 + 8 * hl]);
        v16bf b1 = load_frag(&ldsB[b_row1 * 32 + 8 * hl]);
        acc0 = __builtin_amdgcn_wmma_f32_16x16x32_bf16(false, a, false, b0,
                                                       (short)0, acc0, false, false);
        acc1 = __builtin_amdgcn_wmma_f32_16x16x32_bf16(false, a, false, b1,
                                                       (short)0, acc1, false, false);
        __syncthreads();
    }

    const int m_base = m_blk + (wave & 3) * 16;
    const int n0 = n_blk + (wave >> 2) * 32 + idx16;
    const int n1 = n0 + 16;
#pragma unroll
    for (int r = 0; r < 8; ++r) {
        const int m = m_base + 8 * hl + r;
        Cb[(size_t)m * N + n0] = (bf16_t)acc0[r];
        Cb[(size_t)m * N + n1] = (bf16_t)acc1[r];
    }
}

// ---- f32 -> bf16 conversion with zero padding ----------------------------
__global__ void conv_pad_bf16_kernel(const float* __restrict__ in, bf16_t* __restrict__ out,
                                     int Rin, int Kin, int Kout, int ldin, long long total)
{
    long long idx = (long long)blockIdx.x * 256 + threadIdx.x;
    if (idx >= total) return;
    int r = (int)(idx / Kout);
    int k = (int)(idx % Kout);
    float v = (r < Rin && k < Kin) ? in[(size_t)r * ldin + k] : 0.f;
    out[idx] = (bf16_t)v;
}

__global__ void init_state_kernel(bf16_t* __restrict__ h, float* __restrict__ c)
{
    int i = blockIdx.x * 256 + threadIdx.x;
    if (i < Bdim * Hdim) { h[i] = (bf16_t)0.f; c[i] = 0.f; }
}

// ---- attention score: e[b,s] = sum_h tanh(Hp[b,s,h] + prev[b,h]) * Wsc[h] -
// one wave per (b,s) row; lane covers 16 contiguous h -> 32B coalesced loads.
__global__ void __launch_bounds__(256)
escore_kernel(const bf16_t* __restrict__ Hp, const float* __restrict__ prev,
              const float* __restrict__ Wsc, float* __restrict__ e)
{
    const int lane = threadIdx.x & 31;
    const int wave = threadIdx.x >> 5;
    const int row  = blockIdx.x * 8 + wave;   // row = b*S + s
    const int b    = row >> 9;                // S = 512

    const bf16_t* hp = Hp   + (size_t)row * Hdim + lane * 16;
    const float*  pv = prev + b * Hdim + lane * 16;
    const float*  ws = Wsc  + lane * 16;

    v8bf h0 = *(const v8bf*)(hp);
    v8bf h1 = *(const v8bf*)(hp + 8);
    float acc = 0.f;
#pragma unroll
    for (int i = 0; i < 16; ++i) {
        float x = (float)(i < 8 ? h0[i] : h1[i - 8]) + pv[i];
        acc += tanhf(x) * ws[i];
    }
#pragma unroll
    for (int off = 16; off > 0; off >>= 1) acc += __shfl_down(acc, off, 32);
    if (lane == 0) e[row] = acc;
}

// ---- softmax over S=512, in place ----------------------------------------
__global__ void __launch_bounds__(256)
softmax512_kernel(float* __restrict__ e)
{
    __shared__ float red[256];
    const int b = blockIdx.x, t = threadIdx.x;
    float* eb = e + b * Sdim;
    float v0 = eb[t], v1 = eb[t + 256];
    red[t] = fmaxf(v0, v1); __syncthreads();
    for (int s = 128; s > 0; s >>= 1) { if (t < s) red[t] = fmaxf(red[t], red[t + s]); __syncthreads(); }
    const float m = red[0]; __syncthreads();
    v0 = __expf(v0 - m); v1 = __expf(v1 - m);
    red[t] = v0 + v1; __syncthreads();
    for (int s = 128; s > 0; s >>= 1) { if (t < s) red[t] += red[t + s]; __syncthreads(); }
    const float inv = 1.f / red[0];
    eb[t] = v0 * inv; eb[t + 256] = v1 * inv;
}

// ---- context[b,v] = sum_s alpha[b,s] * batch_H_bf16[b,s,v] ---------------
__global__ void __launch_bounds__(256)
context_kernel(const float* __restrict__ alpha, const bf16_t* __restrict__ bH,
               bf16_t* __restrict__ ctx)
{
    __shared__ float a_s[Sdim];
    const int b = blockIdx.x, t = threadIdx.x;
    a_s[t] = alpha[b * Sdim + t];
    a_s[t + 256] = alpha[b * Sdim + t + 256];
    __syncthreads();
    const bf16_t* base = bH + (size_t)b * Sdim * Vpad;
    for (int v = t; v < Vpad; v += 256) {
        float acc = 0.f;
#pragma unroll 8
        for (int s = 0; s < Sdim; ++s)
            acc += a_s[s] * (float)base[(size_t)s * Vpad + v];
        ctx[b * Vpad + v] = (bf16_t)acc;   // padded v (>=600) is zero -> acc==0
    }
}

// ---- LSTM pointwise + one-hot column gather ------------------------------
__global__ void __launch_bounds__(256)
lstm_kernel(const float* __restrict__ gates, const int* __restrict__ text,
            const float* __restrict__ W_ih, float* __restrict__ c,
            bf16_t* __restrict__ h_bf, bf16_t* __restrict__ hidd, int t)
{
    const int idx = blockIdx.x * 256 + threadIdx.x;  // 0 .. B*H-1
    const int b = idx >> 9;
    const int j = idx & (Hdim - 1);
    const int token = (t == 0) ? 0 : text[b * Tdim + (t - 1)];
    const size_t col = (size_t)Vdim + token;         // one-hot column in W_ih[.,2V]

    float gi = gates[b * G4H + j]            + W_ih[(size_t)(j)        * (2 * Vdim) + col];
    float gf = gates[b * G4H + 512 + j]      + W_ih[(size_t)(512 + j)  * (2 * Vdim) + col];
    float gg = gates[b * G4H + 1024 + j]     + W_ih[(size_t)(1024 + j) * (2 * Vdim) + col];
    float go = gates[b * G4H + 1536 + j]     + W_ih[(size_t)(1536 + j) * (2 * Vdim) + col];

    float si = 1.f / (1.f + __expf(-gi));
    float sf = 1.f / (1.f + __expf(-gf));
    float so = 1.f / (1.f + __expf(-go));
    float cn = sf * c[idx] + si * tanhf(gg);
    float hn = so * tanhf(cn);

    c[idx]    = cn;
    h_bf[idx] = (bf16_t)hn;
    hidd[((size_t)b * Tdim + t) * Hdim + j] = (bf16_t)hn;
}

// ---- final vocab softmax over V=600 --------------------------------------
__global__ void __launch_bounds__(256)
vocab_softmax_kernel(const float* __restrict__ logits, float* __restrict__ out)
{
    __shared__ float red[256];
    const int row = blockIdx.x, t = threadIdx.x;
    const float* lr = logits + (size_t)row * Vdim;
    float m = -3.4e38f;
    for (int v = t; v < Vdim; v += 256) m = fmaxf(m, lr[v]);
    red[t] = m; __syncthreads();
    for (int s = 128; s > 0; s >>= 1) { if (t < s) red[t] = fmaxf(red[t], red[t + s]); __syncthreads(); }
    m = red[0]; __syncthreads();
    float ev[3]; int cnt = 0; float loc = 0.f;
    for (int v = t; v < Vdim; v += 256) { ev[cnt] = __expf(lr[v] - m); loc += ev[cnt]; ++cnt; }
    red[t] = loc; __syncthreads();
    for (int s = 128; s > 0; s >>= 1) { if (t < s) red[t] += red[t + s]; __syncthreads(); }
    const float inv = 1.f / red[0];
    float* orow = out + (size_t)row * Vdim;
    cnt = 0;
    for (int v = t; v < Vdim; v += 256) orow[v] = ev[cnt++] * inv;
}

// ---------------------------------------------------------------------------
extern "C" void kernel_launch(void* const* d_in, const int* in_sizes, int n_in,
                              void* d_out, int out_size, void* d_ws, size_t ws_size,
                              hipStream_t stream)
{
    (void)in_sizes; (void)n_in; (void)out_size; (void)ws_size;

    const float* batch_H = (const float*)d_in[0];   // [B,S,V]
    const int*   text    = (const int*)  d_in[1];   // [B,T]
    const float* W_i2h   = (const float*)d_in[2];   // [H,V]
    const float* W_h2h   = (const float*)d_in[3];   // [H,H]
    const float* b_h2h   = (const float*)d_in[4];   // [H]
    const float* W_score = (const float*)d_in[5];   // [1,H]
    const float* W_ih    = (const float*)d_in[6];   // [4H,2V]
    const float* W_hh    = (const float*)d_in[7];   // [4H,H]
    const float* b_ih    = (const float*)d_in[8];   // [4H]
    const float* b_hh    = (const float*)d_in[9];   // [4H]
    const float* W_gen   = (const float*)d_in[10];  // [V,H]
    const float* b_gen   = (const float*)d_in[11];  // [V]
    float* probs = (float*)d_out;                   // [B,T,V]

    // ---- workspace carve-out (256B aligned) ----
    char* w = (char*)d_ws; size_t off = 0;
    auto alloc = [&](size_t bytes) -> void* {
        void* p = w + off; off = (off + bytes + 255) & ~(size_t)255; return p;
    };
    const int BS = Bdim * Sdim;           // 65536
    const int BT = Bdim * Tdim;           // 15488
    bf16_t* bH_bf   = (bf16_t*)alloc((size_t)BS * Vpad * 2);     // 79.7 MB
    bf16_t* Wi2h_bf = (bf16_t*)alloc((size_t)Hdim * Vpad * 2);
    bf16_t* Wh2h_bf = (bf16_t*)alloc((size_t)Hdim * Hdim * 2);
    bf16_t* Wih_bf  = (bf16_t*)alloc((size_t)G4H  * Vpad * 2);
    bf16_t* Whh_bf  = (bf16_t*)alloc((size_t)G4H  * Hdim * 2);
    bf16_t* Wgen_bf = (bf16_t*)alloc((size_t)640  * Hdim * 2);   // 600 rows -> pad 640
    bf16_t* Hp_bf   = (bf16_t*)alloc((size_t)BS * Hdim * 2);     // 67 MB
    float*  prev    = (float*) alloc((size_t)Bdim * Hdim * 4);
    float*  e_alpha = (float*) alloc((size_t)Bdim * Sdim * 4);
    bf16_t* ctx_bf  = (bf16_t*)alloc((size_t)Bdim * Vpad * 2);
    float*  gates   = (float*) alloc((size_t)Bdim * G4H * 4);
    bf16_t* h_bf    = (bf16_t*)alloc((size_t)Bdim * Hdim * 2);
    float*  c_st    = (float*) alloc((size_t)Bdim * Hdim * 4);
    bf16_t* hidd_bf = (bf16_t*)alloc((size_t)BT * Hdim * 2);
    float*  logits  = (float*) alloc((size_t)BT * Vdim * 4);

    auto conv = [&](const float* in, bf16_t* out, int Rin, int Kin, int Rout, int Kout, int ldin) {
        long long total = (long long)Rout * Kout;
        int grid = (int)((total + 255) / 256);
        conv_pad_bf16_kernel<<<grid, 256, 0, stream>>>(in, out, Rin, Kin, Kout, ldin, total);
    };

    // ---- one-time conversions (zero-padded) ----
    conv(batch_H, bH_bf,  BS,   Vdim, BS,   Vpad, Vdim);
    conv(W_i2h,  Wi2h_bf, Hdim, Vdim, Hdim, Vpad, Vdim);
    conv(W_h2h,  Wh2h_bf, Hdim, Hdim, Hdim, Hdim, Hdim);
    conv(W_ih,   Wih_bf,  G4H,  Vdim, G4H,  Vpad, 2 * Vdim);  // left (context) block only
    conv(W_hh,   Whh_bf,  G4H,  Hdim, G4H,  Hdim, Hdim);
    conv(W_gen,  Wgen_bf, Vdim, Hdim, 640,  Hdim, Hdim);
    init_state_kernel<<<(Bdim * Hdim) / 256, 256, 0, stream>>>(h_bf, c_st);

    // ---- H_proj = batch_H @ W_i2h^T -> bf16 [B*S, H]  (TDM-staged WMMA) ----
    gemm_bf16_tdm_kernel<<<dim3(Hdim / 64, BS / 64), 256, 0, stream>>>(
        bH_bf, Wi2h_bf, Vpad, Hp_bf, Hdim, BS, Hdim);

    // ---- sequential scan over T steps ----
    for (int t = 0; t < Tdim; ++t) {
        // prev_proj = h @ W_h2h^T + b_h2h
        gemm_bf16_kernel<<<dim3(Hdim / 64, Bdim / 64), 256, 0, stream>>>(
            h_bf, Wh2h_bf, Hdim, nullptr, nullptr, 0,
            prev, nullptr, b_h2h, nullptr, Hdim);
        // e = tanh(H_proj + prev) . W_score
        escore_kernel<<<BS / 8, 256, 0, stream>>>(Hp_bf, prev, W_score, e_alpha);
        // alpha = softmax_S(e)  (in place)
        softmax512_kernel<<<Bdim, 256, 0, stream>>>(e_alpha);
        // context = alpha . batch_H  -> bf16 [B, Vpad]
        context_kernel<<<Bdim, 256, 0, stream>>>(e_alpha, bH_bf, ctx_bf);
        // gates = ctx @ W_ih[:, :V]^T + h @ W_hh^T + b_ih + b_hh
        gemm_bf16_kernel<<<dim3(G4H / 64, Bdim / 64), 256, 0, stream>>>(
            ctx_bf, Wih_bf, Vpad, h_bf, Whh_bf, Hdim,
            gates, nullptr, b_ih, b_hh, G4H);
        // pointwise LSTM + one-hot column gather
        lstm_kernel<<<(Bdim * Hdim) / 256, 256, 0, stream>>>(
            gates, text, W_ih, c_st, h_bf, hidd_bf, t);
    }

    // ---- logits = hidden @ W_gen^T + b_gen ; probs = softmax_V ----
    gemm_bf16_kernel<<<dim3((Vdim + 63) / 64, BT / 64), 256, 0, stream>>>(
        hidd_bf, Wgen_bf, Hdim, nullptr, nullptr, 0,
        logits, nullptr, b_gen, nullptr, Vdim);
    vocab_softmax_kernel<<<BT, 256, 0, stream>>>(logits, probs);
}
